// FourierAttention_65317862637748
// MI455X (gfx1250) — matrix operand order repaired and checked by reading, not compiled, not run on hardware
//
#include <hip/hip_runtime.h>
#include <hip/hip_bf16.h>

// FourierAttention (Galerkin linear attention) for MI455X / gfx1250.
// bf16 WMMA (v_wmma_f32_16x16x32_bf16) everywhere, f32 accumulation.
// Weight (B-operand) tiles are staged once per block into LDS in WMMA
// fragment layout (bf16), double-buffered. B fragments are batch-loaded into
// distinct registers so ds_load clauses overlap the WMMA chain instead of
// serializing on s_wait_dscnt 0 per instruction.

typedef __attribute__((ext_vector_type(16))) __bf16 v16bf;
typedef __attribute__((ext_vector_type(8)))  float  v8f;

#define B_      4
#define N_      4096
#define D_      1024
#define H_      8
#define DH_     64
#define INNER_  512
#define BN_     (B_ * N_)               // 16384
#define SEG_    ((size_t)BN_ * INNER_)  // 8388608 elems per q/k/v segment

__device__ __forceinline__ v8f wmma_bf16(v16bf a, v16bf b, v8f c) {
  // (neg_a, A, neg_b, B, c_mod, C, reuse_a, reuse_b)
  return __builtin_amdgcn_wmma_f32_16x16x32_bf16(false, a, false, b, (short)0, c,
                                                 false, false);
}

// Two 16-byte loads -> one fragment of 16 bf16 (works for global or LDS ptrs).
__device__ __forceinline__ v16bf ld_bf16_2x16B(const __bf16* p0, const __bf16* p1) {
  v16bf r;
  uint4* rp = reinterpret_cast<uint4*>(&r);
  rp[0] = *reinterpret_cast<const uint4*>(p0);
  rp[1] = *reinterpret_cast<const uint4*>(p1);
  return r;
}

// Build A fragment (16x32 bf16, ISA layout) from f32 row with on-the-fly cvt.
// lane<16: K {kb8..kb8+7, kb8+16..kb8+23} with kb8=0; lane>=16: kb8=8.
__device__ __forceinline__ v16bf a_frag_from_f32(const float* ap) {
  v16bf a;
#pragma unroll
  for (int j = 0; j < 8; ++j) a[j]     = (__bf16)ap[j];
#pragma unroll
  for (int j = 0; j < 8; ++j) a[8 + j] = (__bf16)ap[16 + j];
  return a;
}

// Stage one 128(N) x 32(K) f32 weight tile -> bf16 LDS in B-fragment order.
// Each of the 256 threads emits exactly one fragment-lane's 16 contiguous
// bf16 (32 B aligned store). Fragment t (t = n>>4), lane L = (n&15)+16*khalf
// lives at lds[((t*32+L)<<4)].
__device__ __forceinline__ void stage_b_tile(const float* __restrict__ w, int ldw,
                                             int n0, int k0, __bf16* lds, int tid) {
  const int n     = tid >> 1;            // 0..127
  const int khalf = tid & 1;             // 0: K 0..7,16..23, 1: K 8..15,24..31
  const float* bp = w + (size_t)(n0 + n) * ldw + k0 + (khalf << 3);
  __bf16 tmp[16];
#pragma unroll
  for (int j = 0; j < 16; ++j) tmp[j] = (__bf16)bp[j + (j & 8)];
  const int t = n >> 4;
  const int L = (n & 15) + (khalf << 4);
  uint4* d4 = reinterpret_cast<uint4*>(lds + ((t * 32 + L) << 4));
  const uint4* s4 = reinterpret_cast<const uint4*>(tmp);
  d4[0] = s4[0];
  d4[1] = s4[1];
}

// ---------------------------------------------------------------------------
// Kernel 1: qkv = x @ W_qkv^T   (M=16384, N=1536, K=1024)
// Block = 8 waves; block tile 128x128; wave tile 16x128 (8x v8f); BK = 32.
// B tile staged in LDS (double-buffered bf16). Writes q/k/v to [3][B,H,N,DH].
// ---------------------------------------------------------------------------
__global__ __launch_bounds__(256)
void qkv_gemm_kernel(const float* __restrict__ x, const float* __restrict__ wqkv,
                     __bf16* __restrict__ qkv) {
  __shared__ __bf16 ldsb[2][128 * 32];   // 2 x 8 KB

  const int tid  = threadIdx.x;
  const int lane = tid & 31;
  const int wv   = tid >> 5;
  const int m0   = blockIdx.x * 128 + wv * 16;
  const int n0   = blockIdx.y * 128;
  const int l15  = lane & 15;
  const int kb8  = (lane >> 4) << 3;

  v8f zero = {};
  v8f acc[8];
#pragma unroll
  for (int t = 0; t < 8; ++t) acc[t] = zero;

  const float* aprow = x + (size_t)(m0 + l15) * D_;

  stage_b_tile(wqkv, D_, n0, 0, ldsb[0], tid);
  __syncthreads();

  int cur = 0;
  for (int k0 = 0; k0 < D_; k0 += 32) {
    if (k0 + 32 < D_) stage_b_tile(wqkv, D_, n0, k0 + 32, ldsb[cur ^ 1], tid);

    const float* ap = aprow + k0 + kb8;
    __builtin_prefetch(ap + 64, 0, 3);             // global_prefetch_b8
    v16bf a = a_frag_from_f32(ap);
    const __bf16* lb = ldsb[cur];

    // Batch-load all 8 B fragments (one ds clause), then run the WMMA chain.
    v16bf bf[8];
#pragma unroll
    for (int t = 0; t < 8; ++t) {
      const __bf16* bt = lb + ((t * 32 + lane) << 4);
      bf[t] = ld_bf16_2x16B(bt, bt + 8);
    }
#pragma unroll
    for (int t = 0; t < 8; ++t) acc[t] = wmma_bf16(a, bf[t], acc[t]);

    __syncthreads();
    cur ^= 1;
  }

  // Epilogue: scatter into [which][B,H,N,DH] bf16.
#pragma unroll
  for (int t = 0; t < 8; ++t) {
    const int c = n0 + t * 16 + l15;
    const int which = c >> 9, rem = c & 511;
    const int h = rem >> 6, dd = rem & 63;
#pragma unroll
    for (int r = 0; r < 8; ++r) {
      const int m = m0 + r + kb8;                  // C layout: M = r + 8*(lane>=16)
      const int bb = m >> 12, n = m & (N_ - 1);
      qkv[(size_t)which * SEG_ + ((size_t)(bb * H_ + h) * N_ + n) * DH_ + dd] =
          (__bf16)acc[t][r];
    }
  }
}

// ---------------------------------------------------------------------------
// Kernel 2: per-head LayerNorm on q and k segments, in place.
// One wave (32 lanes) per 64-element row; f32 statistics via shfl_xor tree.
// ---------------------------------------------------------------------------
__global__ __launch_bounds__(256)
void ln_kernel(__bf16* __restrict__ qkv,
               const float* __restrict__ gq, const float* __restrict__ bq,
               const float* __restrict__ gk, const float* __restrict__ bk) {
  const int lane = threadIdx.x & 31;
  const int wid  = blockIdx.x * (blockDim.x >> 5) + (threadIdx.x >> 5);
  const int nw   = gridDim.x * (blockDim.x >> 5);
  const int ROWS = 2 * B_ * H_ * N_;               // q rows then k rows

  for (int r = wid; r < ROWS; r += nw) {
    const int seg = (r >= B_ * H_ * N_) ? 1 : 0;
    const int rr  = r - seg * (B_ * H_ * N_);
    const int h   = (rr / N_) & (H_ - 1);
    __bf16* row = qkv + (size_t)seg * SEG_ + (size_t)rr * DH_;

    float v0 = (float)row[lane];
    float v1 = (float)row[lane + 32];
    float s = v0 + v1, ss = v0 * v0 + v1 * v1;
#pragma unroll
    for (int m = 16; m >= 1; m >>= 1) {
      s  += __shfl_xor(s,  m, 32);
      ss += __shfl_xor(ss, m, 32);
    }
    const float mean = s * (1.0f / 64.0f);
    const float var  = ss * (1.0f / 64.0f) - mean * mean;
    const float rn   = rsqrtf(var + 1e-5f);
    const float* g  = seg ? gk : gq;
    const float* be = seg ? bk : bq;
    row[lane]      = (__bf16)((v0 - mean) * rn * g[h * DH_ + lane]      + be[h * DH_ + lane]);
    row[lane + 32] = (__bf16)((v1 - mean) * rn * g[h * DH_ + lane + 32] + be[h * DH_ + lane + 32]);
  }
}

// ---------------------------------------------------------------------------
// Kernel 3: kvT[b,h][e][d] = scale * sum_n k_ln[n][d] * v[n][e]
// One block per (b,h). 8 waves split the N=4096 contraction; each wave holds
// the full 64x64 tile (16 accumulators), reduced via LDS ds_add_f32 atomics.
// ---------------------------------------------------------------------------
__global__ __launch_bounds__(256)
void kv_kernel(const __bf16* __restrict__ qkv, __bf16* __restrict__ kvT) {
  __shared__ float kvbuf[DH_ * DH_];
  const int bh = blockIdx.x;                       // 0..31
  const __bf16* kmat = qkv + SEG_     + (size_t)bh * N_ * DH_;
  const __bf16* vmat = qkv + 2 * SEG_ + (size_t)bh * N_ * DH_;

  for (int i = threadIdx.x; i < DH_ * DH_; i += blockDim.x) kvbuf[i] = 0.0f;
  __syncthreads();

  const int lane = threadIdx.x & 31;
  const int wv   = threadIdx.x >> 5;
  const int l15  = lane & 15;
  const int kb8  = (lane >> 4) << 3;
  const int kb16 = (lane >> 4) << 4;

  v8f zero = {};
  v8f acc[4][4];
#pragma unroll
  for (int i = 0; i < 4; ++i)
#pragma unroll
    for (int j = 0; j < 4; ++j) acc[i][j] = zero;

  for (int n0 = wv * 32; n0 < N_; n0 += 256) {
    v16bf bfr[4];                                  // B[k=n][e] = v[n][e]
#pragma unroll
    for (int te = 0; te < 4; ++te) {
      const int e = te * 16 + l15;
      v16bf b;
#pragma unroll
      for (int j = 0; j < 16; ++j) b[j] = vmat[(size_t)(n0 + kb8 + j + (j & 8)) * DH_ + e];
      bfr[te] = b;
    }
#pragma unroll
    for (int td = 0; td < 4; ++td) {
      const int d = td * 16 + l15;                 // A[m=d][k=n] = k_ln[n][d]
      v16bf a;
#pragma unroll
      for (int j = 0; j < 8; ++j) a[j]     = kmat[(size_t)(n0 + kb8 + j) * DH_ + d];
#pragma unroll
      for (int j = 0; j < 8; ++j) a[8 + j] = kmat[(size_t)(n0 + kb8 + 16 + j) * DH_ + d];
#pragma unroll
      for (int te = 0; te < 4; ++te) acc[td][te] = wmma_bf16(a, bfr[te], acc[td][te]);
    }
  }
  // Cross-wave reduction in LDS.
#pragma unroll
  for (int td = 0; td < 4; ++td)
#pragma unroll
    for (int te = 0; te < 4; ++te)
#pragma unroll
      for (int r = 0; r < 8; ++r)
        atomicAdd(&kvbuf[(td * 16 + r + kb8) * DH_ + te * 16 + l15], acc[td][te][r]);
  __syncthreads();

  const float scale = 1.0f / (8.0f * (float)N_);   // 1/(sqrt(DH)*N)
  for (int i = threadIdx.x; i < DH_ * DH_; i += blockDim.x) {
    const int d = i >> 6, e = i & 63;
    kvT[(size_t)bh * (DH_ * DH_) + e * DH_ + d] = (__bf16)(kvbuf[i] * scale);  // transposed
  }
}

// ---------------------------------------------------------------------------
// Kernel 4: attn[b,n, h*64+e] = sum_d q_ln[b,h,n,d] * kvT[b,h][e][d]
// (M=4096 per (b,h), N=64, K=64 -> two WMMA steps.)
// ---------------------------------------------------------------------------
__global__ __launch_bounds__(256)
void attn_kernel(const __bf16* __restrict__ qkv, const __bf16* __restrict__ kvT,
                 __bf16* __restrict__ attn) {
  const int bh = blockIdx.y;
  const int b = bh >> 3, h = bh & 7;
  const int lane = threadIdx.x & 31;
  const int wv   = threadIdx.x >> 5;
  const int m0   = blockIdx.x * 128 + wv * 16;     // row within N_
  const int l15  = lane & 15;
  const int kb8  = (lane >> 4) << 3;
  const int kb16 = (lane >> 4) << 4;

  const __bf16* qmat = qkv + (size_t)bh * N_ * DH_;          // normalized q
  const __bf16* kvm  = kvT + (size_t)bh * (DH_ * DH_);

  v8f zero = {};
  v8f acc[4] = {zero, zero, zero, zero};
  const __bf16* aprow = qmat + (size_t)(m0 + l15) * DH_;
#pragma unroll
  for (int k0 = 0; k0 < DH_; k0 += 32) {
    v16bf a = ld_bf16_2x16B(aprow + k0 + kb8, aprow + k0 + kb8 + 16);
    v16bf bb[4];
#pragma unroll
    for (int te = 0; te < 4; ++te) {
      const __bf16* bp = kvm + (size_t)(te * 16 + l15) * DH_ + k0 + kb8;  // kvT[e][d]
      bb[te] = ld_bf16_2x16B(bp, bp + 16);
    }
#pragma unroll
    for (int te = 0; te < 4; ++te) acc[te] = wmma_bf16(a, bb[te], acc[te]);
  }
#pragma unroll
  for (int te = 0; te < 4; ++te) {
    const int e = te * 16 + l15;
#pragma unroll
    for (int r = 0; r < 8; ++r) {
      const int n = m0 + r + kb8;
      attn[((size_t)(b * N_ + n)) * INNER_ + h * DH_ + e] = (__bf16)acc[te][r];
    }
  }
}

// ---------------------------------------------------------------------------
// Kernel 5: out = attn @ W_out^T + b_out   (M=16384, N=1024, K=512), f32 out.
// Same LDS-staged structure as kernel 1; block tile 128x128.
// ---------------------------------------------------------------------------
__global__ __launch_bounds__(256)
void out_gemm_kernel(const __bf16* __restrict__ attn, const float* __restrict__ wout,
                     const float* __restrict__ bout, float* __restrict__ out) {
  __shared__ __bf16 ldsb[2][128 * 32];   // 2 x 8 KB

  const int tid  = threadIdx.x;
  const int lane = tid & 31;
  const int wv   = tid >> 5;
  const int m0   = blockIdx.x * 128 + wv * 16;
  const int n0   = blockIdx.y * 128;
  const int l15  = lane & 15;
  const int kb8  = (lane >> 4) << 3;

  v8f zero = {};
  v8f acc[8];
#pragma unroll
  for (int t = 0; t < 8; ++t) acc[t] = zero;

  const __bf16* aprow = attn + (size_t)(m0 + l15) * INNER_;

  stage_b_tile(wout, INNER_, n0, 0, ldsb[0], tid);
  __syncthreads();

  int cur = 0;
  for (int k0 = 0; k0 < INNER_; k0 += 32) {
    if (k0 + 32 < INNER_) stage_b_tile(wout, INNER_, n0, k0 + 32, ldsb[cur ^ 1], tid);

    const __bf16* ap = aprow + k0 + kb8;
    __builtin_prefetch(ap + 64, 0, 3);
    v16bf a = ld_bf16_2x16B(ap, ap + 16);
    const __bf16* lb = ldsb[cur];

    v16bf bf[8];
#pragma unroll
    for (int t = 0; t < 8; ++t) {
      const __bf16* bt = lb + ((t * 32 + lane) << 4);
      bf[t] = ld_bf16_2x16B(bt, bt + 8);
    }
#pragma unroll
    for (int t = 0; t < 8; ++t) acc[t] = wmma_bf16(a, bf[t], acc[t]);

    __syncthreads();
    cur ^= 1;
  }

#pragma unroll
  for (int t = 0; t < 8; ++t) {
    const int c = n0 + t * 16 + l15;
    const float bias = bout[c];
#pragma unroll
    for (int r = 0; r < 8; ++r) {
      const int m = m0 + r + kb8;
      out[(size_t)m * D_ + c] = acc[t][r] + bias;
    }
  }
}

// ---------------------------------------------------------------------------
extern "C" void kernel_launch(void* const* d_in, const int* in_sizes, int n_in,
                              void* d_out, int out_size, void* d_ws, size_t ws_size,
                              hipStream_t stream) {
  const float* x     = (const float*)d_in[0];
  const float* W_qkv = (const float*)d_in[1];
  const float* g_q   = (const float*)d_in[2];
  const float* b_q   = (const float*)d_in[3];
  const float* g_k   = (const float*)d_in[4];
  const float* b_k   = (const float*)d_in[5];
  const float* W_out = (const float*)d_in[6];
  const float* b_out = (const float*)d_in[7];
  float* out = (float*)d_out;

  // Workspace layout (bytes):
  //   [0)        qkv bf16  : 3 * SEG_ * 2 = 50,331,648
  //   [qkv_end)  kvT bf16  : 32 * 64*64 * 2 = 262,144
  //   [kvT_end)  attn bf16 : SEG_ * 2 = 16,777,216     (~64.2 MB total)
  char* ws = (char*)d_ws;
  __bf16* qkv  = (__bf16*)ws;
  __bf16* kvT  = (__bf16*)(ws + 3 * SEG_ * sizeof(__bf16));
  __bf16* attn = (__bf16*)(ws + 3 * SEG_ * sizeof(__bf16) +
                           (size_t)32 * DH_ * DH_ * sizeof(__bf16));

  // 1) QKV projection: grid 128 x 12 blocks of 256 threads (tile 128x128).
  qkv_gemm_kernel<<<dim3(BN_ / 128, (3 * INNER_) / 128), 256, 0, stream>>>(x, W_qkv, qkv);

  // 2) Per-head LayerNorm on q and k (in place).
  ln_kernel<<<512, 256, 0, stream>>>(qkv, g_q, b_q, g_k, b_k);

  // 3) kv = K^T V per (b,h), stored transposed + pre-scaled.
  kv_kernel<<<B_ * H_, 256, 0, stream>>>(qkv, kvT);

  // 4) attn = Q @ kv per (b,h).
  attn_kernel<<<dim3(N_ / 128, B_ * H_), 256, 0, stream>>>(qkv, kvT, attn);

  // 5) Output projection + bias.
  out_gemm_kernel<<<dim3(BN_ / 128, D_ / 128), 256, 0, stream>>>(attn, W_out, b_out, out);
}
